// SAttention_49667001811359
// MI455X (gfx1250) — matrix operand order repaired and checked by reading, not compiled, not run on hardware
//
#include <hip/hip_runtime.h>
#include <hip/hip_bf16.h>
#include <math.h>

typedef __attribute__((ext_vector_type(16))) _Float16 v16h;
typedef __attribute__((ext_vector_type(8)))  _Float16 v8h;
typedef __attribute__((ext_vector_type(8)))  float    v8f;
typedef __attribute__((ext_vector_type(4)))  float    v4f;

#define BATCH 16
#define SEQ   1024
#define CH    512
#define NHEAD 8
#define HDIM  64
#define MTOT  (BATCH*SEQ)          /* 16384 */
#define ATT_SCALE 0.125f           /* 64^-0.5 */
#define LOG2E 1.44269504088896f

union V16U { v16h v; v8h h[2]; unsigned u[8]; };

static __device__ inline v16h ld16(const _Float16* p0, const _Float16* p1) {
  V16U t;
  t.h[0] = *(const v8h*)p0;
  t.h[1] = *(const v8h*)p1;
  return t.v;
}

static __device__ inline unsigned pack2(float a, float b) {
  union { _Float16 h[2]; unsigned u; } t;
  t.h[0] = (_Float16)a; t.h[1] = (_Float16)b;
  return t.u;
}

static __device__ inline v8f wmma_f16(v16h a, v16h b, v8f c) {
  return __builtin_amdgcn_wmma_f32_16x16x32_f16(false, a, false, b, (short)0, c, false, false);
}

// ---------------- conversion / transpose helpers ----------------
__global__ void k_cvt_f16(const float* __restrict__ src, _Float16* __restrict__ dst, int n) {
  int i = blockIdx.x * blockDim.x + threadIdx.x;
  if (i < n) dst[i] = (_Float16)src[i];
}

// Wt[n*K + k] = W[k*Ncol + n]   (W is [K, Ncol] row-major -> Wt is [Ncol, K])
__global__ void k_transpose_f16(const float* __restrict__ W, _Float16* __restrict__ Wt,
                                int K, int Ncol) {
  int i = blockIdx.x * blockDim.x + threadIdx.x;
  if (i >= K * Ncol) return;
  int n = i / K, k = i - n * K;
  Wt[i] = (_Float16)W[k * Ncol + n];
}

// ---------------- QKV projection ----------------
// Per wave: one 16-wide c_out tile x 64 rows (4 n-tiles), K-loop over 512.
// A = transposed weight (M = c_out), B = xh^T (N = n).  D(c_out, n): lane holds
// fixed n, c_out contiguous down VGPRs -> 16B f16 stores for q/k.
__global__ void __launch_bounds__(256) k_qkv(
    const _Float16* __restrict__ xh,
    const _Float16* __restrict__ Wqt,   // [512, 512]  (c_out, c_in)
    const _Float16* __restrict__ Wkvt,  // [1024, 512]
    const float* __restrict__ bq,
    const float* __restrict__ bkv,
    _Float16* __restrict__ qh,          // [B, H, N, D] pre-scaled by ATT_SCALE*LOG2E
    _Float16* __restrict__ kh,          // [B, H, N, D]
    _Float16* __restrict__ vt)          // [B, H, D, N]
{
  const int lane = threadIdx.x & 31;
  const int wave = (blockIdx.x * blockDim.x + threadIdx.x) >> 5;
  const int NCT = (3 * CH) / 16;   // 96 column tiles
  const int ct = wave % NCT;
  const int nb = wave / NCT;       // 0..255 (64-row blocks)
  const int c0 = ct * 16;
  const int hl = lane >> 4;
  const int li = lane & 15;
  const int nbase = nb * 64;

  const _Float16* wt;
  int crow;
  if (c0 < CH) { wt = Wqt;  crow = c0; }
  else         { wt = Wkvt; crow = c0 - CH; }
  const _Float16* arow = wt + (size_t)(crow + li) * CH;

  v8f acc[4] = {};
  for (int k0 = 0; k0 < CH; k0 += 32) {
    v16h A = ld16(arow + k0 + 16 * hl, arow + k0 + 16 * hl + 8);
#pragma unroll
    for (int nt = 0; nt < 4; ++nt) {
      const _Float16* xrow = xh + (size_t)(nbase + nt * 16 + li) * CH + k0 + 16 * hl;
      v16h Bv = ld16(xrow, xrow + 8);
      acc[nt] = wmma_f16(A, Bv, acc[nt]);
    }
  }

  if (c0 < CH) {                       // ---- Q ----
    const int h = c0 >> 6;
    const int dbase = (c0 & 63) + 8 * hl;
#pragma unroll
    for (int nt = 0; nt < 4; ++nt) {
      int ng = nbase + nt * 16 + li;
      int b = ng >> 10, nl = ng & 1023;
      _Float16* dst = qh + ((size_t)(b * NHEAD + h) * SEQ + nl) * HDIM + dbase;
      v8h o;
#pragma unroll
      for (int r = 0; r < 8; ++r)
        o[r] = (_Float16)((acc[nt][r] + bq[c0 + 8 * hl + r]) * (ATT_SCALE * LOG2E));
      *(v8h*)dst = o;
    }
  } else if (c0 < 2 * CH) {            // ---- K ----
    const int ck = c0 - CH;
    const int h = ck >> 6;
    const int dbase = (ck & 63) + 8 * hl;
#pragma unroll
    for (int nt = 0; nt < 4; ++nt) {
      int ng = nbase + nt * 16 + li;
      int b = ng >> 10, nl = ng & 1023;
      _Float16* dst = kh + ((size_t)(b * NHEAD + h) * SEQ + nl) * HDIM + dbase;
      v8h o;
#pragma unroll
      for (int r = 0; r < 8; ++r)
        o[r] = (_Float16)(acc[nt][r] + bkv[ck + 8 * hl + r]);
      *(v8h*)dst = o;
    }
  } else {                             // ---- V (transposed store) ----
    const int cv = c0 - 2 * CH;
    const int h = cv >> 6;
#pragma unroll
    for (int nt = 0; nt < 4; ++nt) {
      int ng = nbase + nt * 16 + li;
      int b = ng >> 10, nl = ng & 1023;
#pragma unroll
      for (int r = 0; r < 8; ++r) {
        int d = (cv & 63) + 8 * hl + r;
        vt[((size_t)(b * NHEAD + h) * HDIM + d) * SEQ + nl] =
            (_Float16)(acc[nt][r] + bkv[CH + cv + 8 * hl + r]);
      }
    }
  }
}

// ---------------- flash attention ----------------
// One wave per (b, h, 16-row q block).  S^T = K.Q^T so query idx sits in lanes
// (softmax is per-lane + one shfl_xor(16)); O^T = V^T.P with V pre-transposed.
__global__ void __launch_bounds__(256) k_attn(
    const _Float16* __restrict__ qh,
    const _Float16* __restrict__ kh,
    const _Float16* __restrict__ vt,
    _Float16* __restrict__ mid)        // [B, N, C]
{
  const int lane = threadIdx.x & 31;
  const int wave = (blockIdx.x * blockDim.x + threadIdx.x) >> 5;
  const int nt16 = wave & 63;          // 64 q-tiles per (b,h)
  const int bh = wave >> 6;            // 0..127
  const int hl = lane >> 4, li = lane & 15;
  const int n0 = nt16 * 16;

  const _Float16* qrow = qh + ((size_t)bh * SEQ + (n0 + li)) * HDIM;
  const v16h qb0 = ld16(qrow + 8 * hl, qrow + 16 + 8 * hl);
  const v16h qb1 = ld16(qrow + 32 + 8 * hl, qrow + 48 + 8 * hl);

  const _Float16* kbase = kh + (size_t)bh * SEQ * HDIM;
  const _Float16* vbase = vt + (size_t)bh * HDIM * SEQ;

  v8f oacc[4] = {};
  float mrun = -3.0e38f, srun = 0.f;

  for (int mc = 0; mc < SEQ; mc += 32) {
    v8f s0 = {}, s1 = {};
    {
      const _Float16* kr0 = kbase + (size_t)(mc + li) * HDIM;
      const _Float16* kr1 = kbase + (size_t)(mc + 16 + li) * HDIM;
      v16h A;
      A = ld16(kr0 + 8 * hl, kr0 + 16 + 8 * hl);       s0 = wmma_f16(A, qb0, s0);
      A = ld16(kr0 + 32 + 8 * hl, kr0 + 48 + 8 * hl);  s0 = wmma_f16(A, qb1, s0);
      A = ld16(kr1 + 8 * hl, kr1 + 16 + 8 * hl);       s1 = wmma_f16(A, qb0, s1);
      A = ld16(kr1 + 32 + 8 * hl, kr1 + 48 + 8 * hl);  s1 = wmma_f16(A, qb1, s1);
    }
    // online softmax (logits already scaled by ATT_SCALE*log2e via qh)
    float mloc = s0[0];
#pragma unroll
    for (int r = 0; r < 8; ++r) { mloc = fmaxf(mloc, s0[r]); mloc = fmaxf(mloc, s1[r]); }
    mloc = fmaxf(mloc, __shfl_xor(mloc, 16, 32));
    const float mnew = fmaxf(mrun, mloc);
    const float alpha = exp2f(mrun - mnew);
    float p0[8], p1[8], ls = 0.f;
#pragma unroll
    for (int r = 0; r < 8; ++r) {
      p0[r] = exp2f(s0[r] - mnew);
      p1[r] = exp2f(s1[r] - mnew);
      ls += p0[r] + p1[r];
    }
    ls += __shfl_xor(ls, 16, 32);
    srun = srun * alpha + ls;
    mrun = mnew;
#pragma unroll
    for (int dt = 0; dt < 4; ++dt) oacc[dt] = oacc[dt] * alpha;

    // repack P^T into the B-operand layout (half-wave exchange)
    unsigned a0[4], a1[4];
#pragma unroll
    for (int j = 0; j < 4; ++j) {
      a0[j] = pack2(p0[2 * j], p0[2 * j + 1]);
      a1[j] = pack2(p1[2 * j], p1[2 * j + 1]);
    }
    V16U P;
#pragma unroll
    for (int j = 0; j < 4; ++j) {
      unsigned o0 = (unsigned)__shfl_xor((int)a0[j], 16, 32);
      unsigned o1 = (unsigned)__shfl_xor((int)a1[j], 16, 32);
      P.u[j]     = hl ? o1 : a0[j];
      P.u[4 + j] = hl ? a1[j] : o0;
    }
#pragma unroll
    for (int dt = 0; dt < 4; ++dt) {
      const _Float16* vrow = vbase + (size_t)(dt * 16 + li) * SEQ + mc;
      v16h A2 = ld16(vrow + 16 * hl, vrow + 16 * hl + 8);
      oacc[dt] = wmma_f16(A2, P.v, oacc[dt]);
    }
  }

  const float inv = 1.0f / srun;
  const int b = bh >> 3, h = bh & 7;
  const int n = n0 + li;
  _Float16* mrow = mid + ((size_t)b * SEQ + n) * CH + h * HDIM + 8 * hl;
#pragma unroll
  for (int dt = 0; dt < 4; ++dt) {
    v8h o;
#pragma unroll
    for (int r = 0; r < 8; ++r) o[r] = (_Float16)(oacc[dt][r] * inv);
    *(v8h*)(mrow + dt * 16) = o;
  }
}

// ---------------- depthwise 3x3 conv on V, added into mid ----------------
__global__ void k_conv_add(const _Float16* __restrict__ vt,   // [B, C, 1024] == NCHW 32x32
                           const float* __restrict__ cw,      // [C, 1, 3, 3]
                           const float* __restrict__ cb,
                           _Float16* __restrict__ mid) {
  int i = blockIdx.x * blockDim.x + threadIdx.x;
  if (i >= BATCH * CH * SEQ) return;
  const int n = i & (SEQ - 1);
  const int c = (i >> 10) & (CH - 1);
  const int b = i >> 19;
  const int y = n >> 5, x = n & 31;
  const _Float16* vrow = vt + (size_t)(b * CH + c) * SEQ;
  float acc = cb[c];
#pragma unroll
  for (int ky = 0; ky < 3; ++ky) {
    const int yy = y + ky - 1;
    if (yy < 0 || yy > 31) continue;
#pragma unroll
    for (int kx = 0; kx < 3; ++kx) {
      const int xx = x + kx - 1;
      if (xx < 0 || xx > 31) continue;
      acc += cw[c * 9 + ky * 3 + kx] * (float)vrow[yy * 32 + xx];
    }
  }
  const size_t mo = ((size_t)b * SEQ + n) * CH + c;
  mid[mo] = (_Float16)((float)mid[mo] + acc);
}

// ---------------- output projection with transposed store ----------------
// A = mid (M = n), B = Wp^T (N = c_out).  D(n, c_out): lane holds fixed c_out,
// n contiguous down VGPRs -> float4 stores straight into out[b, c, n].
__global__ void __launch_bounds__(256) k_proj(
    const _Float16* __restrict__ mid,
    const _Float16* __restrict__ Wpt,   // [c_out, c_in]
    const float* __restrict__ bp,
    float* __restrict__ out)            // [B, C, N]
{
  const int lane = threadIdx.x & 31;
  const int wave = (blockIdx.x * blockDim.x + threadIdx.x) >> 5;
  const int NCT = CH / 16;              // 32
  const int ct = wave % NCT;
  const int nb = wave / NCT;            // 0..255
  const int c0 = ct * 16;
  const int hl = lane >> 4, li = lane & 15;
  const int nbase = nb * 64;

  const _Float16* brow = Wpt + (size_t)(c0 + li) * CH;
  v8f acc[4] = {};
  for (int k0 = 0; k0 < CH; k0 += 32) {
    v16h Bv = ld16(brow + k0 + 8 * hl, brow + k0 + 16 + 8 * hl);
#pragma unroll
    for (int nt = 0; nt < 4; ++nt) {
      const _Float16* arow = mid + (size_t)(nbase + nt * 16 + li) * CH;
      v16h A = ld16(arow + k0 + 8 * hl, arow + k0 + 16 + 8 * hl);
      acc[nt] = wmma_f16(A, Bv, acc[nt]);
    }
  }
  const float bias = bp[c0 + li];
  const int b = nbase >> 10;
  const int nl = nbase & 1023;
  float* obase = out + ((size_t)b * CH + (c0 + li)) * SEQ;
#pragma unroll
  for (int nt = 0; nt < 4; ++nt) {
    float* p = obase + nl + nt * 16 + 8 * hl;
    v4f lo = {acc[nt][0] + bias, acc[nt][1] + bias, acc[nt][2] + bias, acc[nt][3] + bias};
    v4f hi = {acc[nt][4] + bias, acc[nt][5] + bias, acc[nt][6] + bias, acc[nt][7] + bias};
    *(v4f*)p = lo;
    *(v4f*)(p + 4) = hi;
  }
}

// ---------------- host side ----------------
extern "C" void kernel_launch(void* const* d_in, const int* in_sizes, int n_in,
                              void* d_out, int out_size, void* d_ws, size_t ws_size,
                              hipStream_t stream) {
  (void)in_sizes; (void)n_in; (void)out_size; (void)ws_size;
  const float* x   = (const float*)d_in[0];
  const float* Wq  = (const float*)d_in[1];
  const float* bq  = (const float*)d_in[2];
  const float* Wkv = (const float*)d_in[3];
  const float* bkv = (const float*)d_in[4];
  const float* cw  = (const float*)d_in[5];
  const float* cb  = (const float*)d_in[6];
  const float* Wp  = (const float*)d_in[7];
  const float* bp  = (const float*)d_in[8];
  float* out = (float*)d_out;

  char* ws = (char*)d_ws;
  const size_t SZ_XH  = (size_t)MTOT * CH * 2;        // 16 MiB
  const size_t SZ_WQT = (size_t)CH * CH * 2;          // 0.5 MiB
  const size_t SZ_WKVT= (size_t)2 * CH * CH * 2;      // 1 MiB
  const size_t SZ_WPT = (size_t)CH * CH * 2;          // 0.5 MiB
  const size_t SZ_T   = (size_t)MTOT * CH * 2;        // 16 MiB each
  _Float16* xh   = (_Float16*)(ws);
  _Float16* Wqt  = (_Float16*)(ws + SZ_XH);
  _Float16* Wkvt = (_Float16*)(ws + SZ_XH + SZ_WQT);
  _Float16* Wpt  = (_Float16*)(ws + SZ_XH + SZ_WQT + SZ_WKVT);
  _Float16* qh   = (_Float16*)(ws + SZ_XH + SZ_WQT + SZ_WKVT + SZ_WPT);
  _Float16* kh   = (_Float16*)((char*)qh + SZ_T);
  _Float16* vt   = (_Float16*)((char*)kh + SZ_T);
  _Float16* mid  = (_Float16*)((char*)vt + SZ_T);

  // 1. convert x to f16, transpose weights to [c_out, c_in] f16
  k_cvt_f16<<<(MTOT * CH) / 256, 256, 0, stream>>>(x, xh, MTOT * CH);
  k_transpose_f16<<<(CH * CH + 255) / 256, 256, 0, stream>>>(Wq, Wqt, CH, CH);
  k_transpose_f16<<<(CH * 2 * CH + 255) / 256, 256, 0, stream>>>(Wkv, Wkvt, CH, 2 * CH);
  k_transpose_f16<<<(CH * CH + 255) / 256, 256, 0, stream>>>(Wp, Wpt, CH, CH);

  // 2. QKV projection: 96 col-tiles x 256 row-blocks = 24576 waves
  k_qkv<<<3072, 256, 0, stream>>>(xh, Wqt, Wkvt, bq, bkv, qh, kh, vt);

  // 3. attention: 128 (b,h) x 64 q-tiles = 8192 waves
  k_attn<<<1024, 256, 0, stream>>>(qh, kh, vt, mid);

  // 4. depthwise conv + add
  k_conv_add<<<(BATCH * CH * SEQ) / 256, 256, 0, stream>>>(vt, cw, cb, mid);

  // 5. output projection (writes transposed [B, C, N] directly)
  k_proj<<<1024, 256, 0, stream>>>(mid, Wpt, bp, out);
}
